// NetVLAD_11209864642695
// MI455X (gfx1250) — matrix operand order repaired and checked by reading, not compiled, not run on hardware
//
#include <hip/hip_runtime.h>
#include <hip/hip_bf16.h>

typedef __attribute__((ext_vector_type(16))) _Float16 v16h;
typedef __attribute__((ext_vector_type(8)))  _Float16 v8h;
typedef __attribute__((ext_vector_type(4)))  _Float16 v4h;
typedef __attribute__((ext_vector_type(8)))  float    v8f;

union FragA { v16h v; v8h h[2]; };

#define WMMA_F16(A, B, C) \
    __builtin_amdgcn_wmma_f32_16x16x32_f16(false, (A), false, (B), (short)0, (C), false, false)

#define NN 32
#define CC 512
#define SS 3600
#define SP 3616
#define KK 64
#define EPSN 1e-12f

// ---------------------------------------------------------------------------
// Zero pad regions + accumulators (runs every call: deterministic)
// ---------------------------------------------------------------------------
__global__ void k_init(_Float16* __restrict__ xn_cs, _Float16* __restrict__ a_ws,
                       float* __restrict__ asum, float* __restrict__ gsum) {
    const int pad = SP - SS;                 // 16
    const int t1 = NN * CC * pad;            // xn_cs pad elements
    const int t2 = NN * KK * pad;            // a_ws  pad elements
    int idx = blockIdx.x * blockDim.x + threadIdx.x;
    if (idx < t1) {
        int p = idx % pad;
        int c = (idx / pad) % CC;
        int n = idx / (pad * CC);
        xn_cs[(size_t)n * CC * SP + (size_t)c * SP + SS + p] = (_Float16)0.0f;
    } else if (idx < t1 + t2) {
        int j = idx - t1;
        int p = j % pad;
        int k = (j / pad) % KK;
        int n = j / (pad * KK);
        a_ws[(size_t)n * KK * SP + (size_t)k * SP + SS + p] = (_Float16)0.0f;
    } else if (idx < t1 + t2 + NN * KK) {
        asum[idx - t1 - t2] = 0.0f;
    } else if (idx < t1 + t2 + NN * KK + NN) {
        gsum[idx - t1 - t2 - NN * KK] = 0.0f;
    }
}

// ---------------------------------------------------------------------------
// Per-pixel L2 normalize over C; emit xn in [n][s][c] and [n][c][Sp] f16
// ---------------------------------------------------------------------------
__global__ void k_normalize(const float* __restrict__ x,
                            _Float16* __restrict__ xn_sc,
                            _Float16* __restrict__ xn_cs) {
    int n = blockIdx.y;
    int s = blockIdx.x * blockDim.x + threadIdx.x;
    if (s >= SS) return;
    const float* xp = x + (size_t)n * CC * SS + s;
    float acc = 0.0f;
    #pragma unroll 8
    for (int c = 0; c < CC; ++c) {
        float v = xp[(size_t)c * SS];
        acc += v * v;
    }
    float scale = 1.0f / fmaxf(sqrtf(acc), EPSN);
    _Float16* psc = xn_sc + (size_t)n * SS * CC + (size_t)s * CC;
    _Float16* pcs = xn_cs + (size_t)n * CC * SP + s;
    for (int c0 = 0; c0 < CC; c0 += 8) {
        v8h pk;
        #pragma unroll
        for (int j = 0; j < 8; ++j) {
            float v = xp[(size_t)(c0 + j) * SS] * scale;
            _Float16 h = (_Float16)v;
            pk[j] = h;
            pcs[(size_t)(c0 + j) * SP] = h;   // coalesced across threads per c
        }
        *(v8h*)(psc + c0) = pk;               // one 16B store per 8 channels
    }
}

// ---------------------------------------------------------------------------
// GEMM1 (logits = xn^T * W^T) fused with softmax over k.
// W staged in LDS (64KB, converted f32->f16 in-place during staging).
// One wave per 16-pixel tile; D tiles are 16(s) x 16(k), 4 k-tiles.
// A stream double-buffered from global; B fragments from LDS.
// ---------------------------------------------------------------------------
__global__ void k_gemm1_softmax(const _Float16* __restrict__ xn_sc,
                                const float* __restrict__ conv_w,
                                _Float16* __restrict__ a_ws,
                                float* __restrict__ asum) {
    __shared__ __align__(16) _Float16 wlds[KK * CC];   // 64 KB

    const int ntiles = SS / 16;                        // 225
    int n     = blockIdx.y;
    int wv    = threadIdx.x >> 5;
    int lane  = threadIdx.x & 31;
    int stile = blockIdx.x * (blockDim.x >> 5) + wv;   // wave-uniform
    int hi = lane >> 4;
    int lm = lane & 15;

    // Stage + convert conv_w (f32) -> LDS (f16), all threads participate.
    {
        const float4* wsrc = (const float4*)conv_w;
        v4h* wdst = (v4h*)wlds;
        #pragma unroll
        for (int i = threadIdx.x; i < KK * CC / 4; i += 256) {
            float4 f = wsrc[i];
            v4h h;
            h[0] = (_Float16)f.x; h[1] = (_Float16)f.y;
            h[2] = (_Float16)f.z; h[3] = (_Float16)f.w;
            wdst[i] = h;
        }
    }
    __syncthreads();

    if (stile < ntiles) {
        v8f acc0 = {}, acc1 = {}, acc2 = {}, acc3 = {};
        const _Float16* arow =
            xn_sc + (size_t)n * SS * CC + (size_t)(stile * 16 + lm) * CC;
        const _Float16* wrow = wlds + (size_t)lm * CC + 16 * hi;

        FragA a;
        a.h[0] = *(const v8h*)(arow + 16 * hi);
        a.h[1] = *(const v8h*)(arow + 8 + 16 * hi);

        #pragma unroll
        for (int cc = 0; cc < CC; cc += 32) {
            FragA an;
            if (cc + 32 < CC) {   // static after unroll: prefetch next A chunk
                an.h[0] = *(const v8h*)(arow + cc + 32 + 16 * hi);
                an.h[1] = *(const v8h*)(arow + cc + 40 + 16 * hi);
            }
            FragA b;
            b.h[0] = *(const v8h*)(wrow + cc);                      // ds_load_b128
            b.h[1] = *(const v8h*)(wrow + cc + 8);
            acc0 = WMMA_F16(a.v, b.v, acc0);
            b.h[0] = *(const v8h*)(wrow + 16 * CC + cc);
            b.h[1] = *(const v8h*)(wrow + 16 * CC + cc + 8);
            acc1 = WMMA_F16(a.v, b.v, acc1);
            b.h[0] = *(const v8h*)(wrow + 32 * CC + cc);
            b.h[1] = *(const v8h*)(wrow + 32 * CC + cc + 8);
            acc2 = WMMA_F16(a.v, b.v, acc2);
            b.h[0] = *(const v8h*)(wrow + 48 * CC + cc);
            b.h[1] = *(const v8h*)(wrow + 48 * CC + cc + 8);
            acc3 = WMMA_F16(a.v, b.v, acc3);
            if (cc + 32 < CC) a = an;
        }

        // Softmax over k (64) per row: cross-lane reduce within 16-lane groups
        float att0[8], att1[8], att2[8], att3[8];
        #pragma unroll
        for (int r = 0; r < 8; ++r) {
            float m = fmaxf(fmaxf(acc0[r], acc1[r]), fmaxf(acc2[r], acc3[r]));
            #pragma unroll
            for (int msk = 1; msk < 16; msk <<= 1)
                m = fmaxf(m, __shfl_xor(m, msk, 32));
            float e0 = __expf(acc0[r] - m);
            float e1 = __expf(acc1[r] - m);
            float e2 = __expf(acc2[r] - m);
            float e3 = __expf(acc3[r] - m);
            float ssum = e0 + e1 + e2 + e3;
            #pragma unroll
            for (int msk = 1; msk < 16; msk <<= 1)
                ssum += __shfl_xor(ssum, msk, 32);
            float inv = 1.0f / ssum;
            att0[r] = e0 * inv; att1[r] = e1 * inv;
            att2[r] = e2 * inv; att3[r] = e3 * inv;
        }

        // Store a (f16) in [n][k][Sp]; each lane: 8 consecutive s per tile.
        _Float16* abase = a_ws + (size_t)n * KK * SP;
        size_t soff = (size_t)stile * 16 + 8 * hi;
        {
            v8h pk; float part = 0.0f;
            #pragma unroll
            for (int r = 0; r < 8; ++r) { pk[r] = (_Float16)att0[r]; part += att0[r]; }
            *(v8h*)(abase + (size_t)(0 * 16 + lm) * SP + soff) = pk;
            atomicAdd(asum + n * KK + 0 * 16 + lm, part);
        }
        {
            v8h pk; float part = 0.0f;
            #pragma unroll
            for (int r = 0; r < 8; ++r) { pk[r] = (_Float16)att1[r]; part += att1[r]; }
            *(v8h*)(abase + (size_t)(1 * 16 + lm) * SP + soff) = pk;
            atomicAdd(asum + n * KK + 1 * 16 + lm, part);
        }
        {
            v8h pk; float part = 0.0f;
            #pragma unroll
            for (int r = 0; r < 8; ++r) { pk[r] = (_Float16)att2[r]; part += att2[r]; }
            *(v8h*)(abase + (size_t)(2 * 16 + lm) * SP + soff) = pk;
            atomicAdd(asum + n * KK + 2 * 16 + lm, part);
        }
        {
            v8h pk; float part = 0.0f;
            #pragma unroll
            for (int r = 0; r < 8; ++r) { pk[r] = (_Float16)att3[r]; part += att3[r]; }
            *(v8h*)(abase + (size_t)(3 * 16 + lm) * SP + soff) = pk;
            atomicAdd(asum + n * KK + 3 * 16 + lm, part);
        }
    }
}

// ---------------------------------------------------------------------------
// GEMM2: agg[n,k,c] = sum_s a[k,s] * xn[c,s].
// One wave per 16(k) x 32(c) tile pair: A reused across 2 WMMAs,
// fragments double-buffered across s-chunks.
// ---------------------------------------------------------------------------
__global__ void k_gemm2(const _Float16* __restrict__ a_ws,
                        const _Float16* __restrict__ xn_cs,
                        float* __restrict__ agg) {
    int n    = blockIdx.y;
    int wv   = threadIdx.x >> 5;
    int lane = threadIdx.x & 31;
    int task = blockIdx.x * (blockDim.x >> 5) + wv;   // 0..63
    int kt  = task >> 4;                               // 0..3
    int ctg = task & 15;                               // 0..15 (pair of c-tiles)
    int hi = lane >> 4;
    int lm = lane & 15;

    const _Float16* ap  = a_ws + (size_t)n * KK * SP + (size_t)(kt * 16 + lm) * SP;
    const _Float16* bp0 = xn_cs + (size_t)n * CC * SP + (size_t)(ctg * 32 + lm) * SP;
    const _Float16* bp1 = bp0 + (size_t)16 * SP;

    FragA a, b0, b1;
    a.h[0]  = *(const v8h*)(ap + 16 * hi);
    a.h[1]  = *(const v8h*)(ap + 16 * hi + 8);
    b0.h[0] = *(const v8h*)(bp0 + 16 * hi);
    b0.h[1] = *(const v8h*)(bp0 + 16 * hi + 8);
    b1.h[0] = *(const v8h*)(bp1 + 16 * hi);
    b1.h[1] = *(const v8h*)(bp1 + 16 * hi + 8);

    v8f acc0 = {}, acc1 = {};
    #pragma unroll 2
    for (int sc = 32; sc < SP; sc += 32) {
        FragA na, nb0, nb1;
        na.h[0]  = *(const v8h*)(ap + sc + 16 * hi);
        na.h[1]  = *(const v8h*)(ap + sc + 16 * hi + 8);
        nb0.h[0] = *(const v8h*)(bp0 + sc + 16 * hi);
        nb0.h[1] = *(const v8h*)(bp0 + sc + 16 * hi + 8);
        nb1.h[0] = *(const v8h*)(bp1 + sc + 16 * hi);
        nb1.h[1] = *(const v8h*)(bp1 + sc + 16 * hi + 8);
        __builtin_prefetch(bp0 + sc + 128, 0, 3);
        __builtin_prefetch(bp1 + sc + 128, 0, 3);
        acc0 = WMMA_F16(a.v, b0.v, acc0);
        acc1 = WMMA_F16(a.v, b1.v, acc1);
        a = na; b0 = nb0; b1 = nb1;
    }
    acc0 = WMMA_F16(a.v, b0.v, acc0);
    acc1 = WMMA_F16(a.v, b1.v, acc1);

    float* op = agg + (size_t)n * KK * CC + (size_t)(kt * 16 + 8 * hi) * CC
              + (size_t)ctg * 32 + lm;
    #pragma unroll
    for (int r = 0; r < 8; ++r) {
        op[(size_t)r * CC] = acc0[r];
        op[(size_t)r * CC + 16] = acc1[r];
    }
}

// ---------------------------------------------------------------------------
// vlad = agg - asum*centroid; intra L2-normalize over C per (n,k);
// accumulate per-n global sum of squares of normalized rows.
// ---------------------------------------------------------------------------
__global__ void k_vlad(const float* __restrict__ agg,
                       const float* __restrict__ asum,
                       const float* __restrict__ centroids,
                       float* __restrict__ out,
                       float* __restrict__ gsum) {
    int n = blockIdx.y;
    int k = blockIdx.x;
    int tid = threadIdx.x;                      // 128 threads
    float sk = asum[n * KK + k];
    const float* ag = agg + (size_t)n * KK * CC + (size_t)k * CC;
    const float* ce = centroids + (size_t)k * CC;

    float v[4];
    float ss = 0.0f;
    #pragma unroll
    for (int i = 0; i < 4; ++i) {
        int c = tid + i * 128;
        float t = ag[c] - sk * ce[c];
        v[i] = t;
        ss += t * t;
    }
    #pragma unroll
    for (int msk = 1; msk < 32; msk <<= 1)
        ss += __shfl_xor(ss, msk, 32);
    __shared__ float red[4];
    if ((tid & 31) == 0) red[tid >> 5] = ss;
    __syncthreads();
    float tot = red[0] + red[1] + red[2] + red[3];
    float scale = 1.0f / fmaxf(sqrtf(tot), EPSN);

    float* op = out + (size_t)n * KK * CC + (size_t)k * CC;
    #pragma unroll
    for (int i = 0; i < 4; ++i)
        op[tid + i * 128] = v[i] * scale;

    if (tid == 0)
        atomicAdd(gsum + n, tot * scale * scale);
}

// ---------------------------------------------------------------------------
// Global L2 normalization over flattened (K*C) per image.
// ---------------------------------------------------------------------------
__global__ void k_gnorm(float* __restrict__ out, const float* __restrict__ gsum) {
    int n = blockIdx.y;
    int i = blockIdx.x * blockDim.x + threadIdx.x;
    const int per_n = KK * CC;
    if (i < per_n) {
        float g = 1.0f / fmaxf(sqrtf(gsum[n]), EPSN);
        out[(size_t)n * per_n + i] *= g;
    }
}

// ---------------------------------------------------------------------------
extern "C" void kernel_launch(void* const* d_in, const int* in_sizes, int n_in,
                              void* d_out, int out_size, void* d_ws, size_t ws_size,
                              hipStream_t stream) {
    const float* x         = (const float*)d_in[0];   // [N,C,H,W]
    const float* conv_w    = (const float*)d_in[1];   // [K,C]
    const float* centroids = (const float*)d_in[2];   // [K,C]
    float* out = (float*)d_out;                       // [N, K*C]

    char* p = (char*)d_ws;
    _Float16* xn_sc = (_Float16*)p; p += (size_t)NN * SS * CC * sizeof(_Float16);
    _Float16* xn_cs = (_Float16*)p; p += (size_t)NN * CC * SP * sizeof(_Float16);
    _Float16* a_ws  = (_Float16*)p; p += (size_t)NN * KK * SP * sizeof(_Float16);
    float*    asum  = (float*)p;    p += (size_t)NN * KK * sizeof(float);
    float*    agg   = (float*)p;    p += (size_t)NN * KK * CC * sizeof(float);
    float*    gsum  = (float*)p;    p += (size_t)NN * sizeof(float);

    // 1) init pads + accumulators
    {
        int total = NN * CC * (SP - SS) + NN * KK * (SP - SS) + NN * KK + NN;
        int blocks = (total + 255) / 256;
        k_init<<<blocks, 256, 0, stream>>>(xn_cs, a_ws, asum, gsum);
    }
    // 2) normalize + dual-layout pack
    {
        dim3 grid((SS + 255) / 256, NN);
        k_normalize<<<grid, 256, 0, stream>>>(x, xn_sc, xn_cs);
    }
    // 3) GEMM1 + softmax (225 s-tiles, 8 waves/block)
    {
        dim3 grid((225 + 7) / 8, NN);
        k_gemm1_softmax<<<grid, 256, 0, stream>>>(xn_sc, conv_w, a_ws, asum);
    }
    // 4) GEMM2 (64 wave-tasks per image, 8 waves/block)
    {
        dim3 grid(8, NN);
        k_gemm2<<<grid, 256, 0, stream>>>(a_ws, xn_cs, agg);
    }
    // 5) vlad + intra-normalization
    {
        dim3 grid(KK, NN);
        k_vlad<<<grid, 128, 0, stream>>>(agg, asum, centroids, out, gsum);
    }
    // 6) global normalization
    {
        dim3 grid((KK * CC + 255) / 256, NN);
        k_gnorm<<<grid, 256, 0, stream>>>(out, gsum);
    }
}
